// Decoder_2886218023649
// MI455X (gfx1250) — matrix operand order repaired and checked by reading, not compiled, not run on hardware
//
#include <hip/hip_runtime.h>
#include <math.h>

// Problem constants (match reference)
constexpr int kV = 32000;
constexpr int kH = 512;
constexpr int kB = 16;
constexpr int kT = 64;
constexpr int kF = 3;
constexpr int kD = 4 * kH;           // 2048
constexpr int kM = kB * kT;          // 1024 rows (m = b*T + t)

typedef __attribute__((ext_vector_type(16))) __bf16 bf16x16;
typedef __attribute__((ext_vector_type(8)))  float  f32x8;

__device__ __forceinline__ int lane_id() { return threadIdx.x & 31; }

// pack two float4 (8 fp32) into bf16 halves [base..base+7] of a fragment
__device__ __forceinline__ void cvt8(bf16x16& d, int base, float4 a, float4 b) {
  d[base + 0] = (__bf16)a.x; d[base + 1] = (__bf16)a.y;
  d[base + 2] = (__bf16)a.z; d[base + 3] = (__bf16)a.w;
  d[base + 4] = (__bf16)b.x; d[base + 5] = (__bf16)b.y;
  d[base + 6] = (__bf16)b.z; d[base + 7] = (__bf16)b.w;
}

// ---------------- WMMA fragment helpers --------------------------------------
// A fragment (16x32), fp32 source: src points at tile row 0, leading dim lda.
// Layout: lane L holds row L%16; halves 0..15 = K (L/16)*16 .. +15 (contiguous).
__device__ __forceinline__ bf16x16 load_a(const float* __restrict__ src, int lda, int k0) {
  const int l = lane_id();
  const float* p = src + (size_t)(l & 15) * lda + k0 + ((l >> 4) * 16);
  float4 q0 = *(const float4*)(p);
  float4 q1 = *(const float4*)(p + 4);
  float4 q2 = *(const float4*)(p + 8);
  float4 q3 = *(const float4*)(p + 12);
  bf16x16 a;
  cvt8(a, 0, q0, q1);
  cvt8(a, 8, q2, q3);
  return a;
}

// B fragment (32x16), B[k][n] = W[n][k], W row-major fp32 [N, ldw] ([out,in] weights).
// Lane L holds column L%16; K (L/16)*16 .. +15 -> 16 contiguous floats of one weight row.
__device__ __forceinline__ bf16x16 load_bT(const float* __restrict__ W, int ldw, int n0, int k0) {
  const int l = lane_id();
  const float* p = W + (size_t)(n0 + (l & 15)) * ldw + k0 + ((l >> 4) * 16);
  float4 q0 = *(const float4*)(p);
  float4 q1 = *(const float4*)(p + 4);
  float4 q2 = *(const float4*)(p + 8);
  float4 q3 = *(const float4*)(p + 12);
  bf16x16 b;
  cvt8(b, 0, q0, q1);
  cvt8(b, 8, q2, q3);
  return b;
}

// B fragment (32x16), B[k][n] = W[n][k], W row-major bf16 [N, ldw] (pre-transposed
// weights): one contiguous 32-byte run per lane -> two global_load_b128, no cvt.
__device__ __forceinline__ bf16x16 load_bT_bf16(const __bf16* __restrict__ W, int ldw,
                                                int n0, int k0) {
  const int l = lane_id();
  const __bf16* p = W + (size_t)(n0 + (l & 15)) * ldw + k0 + ((l >> 4) * 16);
  return *(const bf16x16*)p;
}

__device__ __forceinline__ f32x8 wmma_bf16(bf16x16 a, bf16x16 b, f32x8 c) {
  return __builtin_amdgcn_wmma_f32_16x16x32_bf16(false, a, false, b, (short)0, c, false, false);
}

// D fragment store: dst points at tile row 0 (leading dim ldd), columns n0..n0+15.
__device__ __forceinline__ void store_d(float* __restrict__ dst, int ldd, int n0, f32x8 d,
                                        const float* __restrict__ bias, bool relu) {
  const int l = lane_id();
  const int n = n0 + (l & 15);
  const int mb = (l >> 4) * 8;
  const float bv = bias ? bias[n] : 0.0f;
#pragma unroll
  for (int r = 0; r < 8; ++r) {
    float v = d[r] + bv;
    if (relu) v = fmaxf(v, 0.0f);
    dst[(size_t)(mb + r) * ldd + n] = v;
  }
}

// ---------------- Kernel 0: weight prep (transpose fp32 [R,C] -> bf16 [C,R]) --
// Coalesced reads; one-time pass so strided writes are acceptable.
__global__ void transpose_to_bf16_kernel(const float* __restrict__ in, __bf16* __restrict__ out,
                                         int R, int C) {
  const size_t per_f = (size_t)R * C;
  const float* inf = in + blockIdx.y * per_f;
  __bf16* outf = out + blockIdx.y * per_f;
  const size_t i = (size_t)blockIdx.x * blockDim.x + threadIdx.x;
  if (i < per_f) {
    const int r = (int)(i / C), c = (int)(i % C);
    outf[(size_t)c * R + r] = (__bf16)inf[i];
  }
}

// ---------------- Kernel 1: embedding gather + projection ---------------------
// x[m, n] = sum_k emb_cat[m,k] * emb_proj_w[n,k] + emb_proj_b[n],  m = b*T + t
__global__ void embed_proj_kernel(const int* __restrict__ tgt, const int* __restrict__ bosp,
                                  const float* __restrict__ emb, const float* __restrict__ Wp,
                                  const float* __restrict__ bp, float* __restrict__ x) {
  const int tile = blockIdx.x * 8 + (threadIdx.x >> 5);
  const int NT = kH / 16;                 // 32
  const int mt = tile / NT, nt = tile % NT;
  const int m0 = mt * 16, n0 = nt * 16;
  const int l = lane_id();
  const int m = m0 + (l & 15);
  const int bos = bosp[0];
  f32x8 acc = {};
  for (int k0 = 0; k0 < kF * kH; k0 += 32) {
    const int f = k0 / kH;                // constant over this 32-wide K tile
    const int tok = ((m % kT) == 0) ? bos : tgt[(m - 1) * kF + f];
    const float* erow = emb + (size_t)tok * kH + (k0 % kH) + ((l >> 4) * 16);
    float4 q0 = *(const float4*)(erow);
    float4 q1 = *(const float4*)(erow + 4);
    float4 q2 = *(const float4*)(erow + 8);
    float4 q3 = *(const float4*)(erow + 12);
    bf16x16 a;
    cvt8(a, 0, q0, q1);
    cvt8(a, 8, q2, q3);
    bf16x16 b = load_bT(Wp, kF * kH, n0, k0);
    acc = wmma_bf16(a, b, acc);
  }
  store_d(x + (size_t)m0 * kH, kH, n0, acc, bp, false);
}

// ---------------- Kernel 2: one LSTM step for one layer ----------------------
// Block = 4 waves (one per gate i,f,g,o), handles a 16-wide hidden slice.
__global__ void lstm_step_kernel(const float* __restrict__ x_t, int x_stride,
                                 const float* __restrict__ wih, const float* __restrict__ whh,
                                 const float* __restrict__ bih, const float* __restrict__ bhh,
                                 const float* __restrict__ h_in, float* __restrict__ h_out,
                                 float* __restrict__ c_st,
                                 float* __restrict__ y_t, int y_stride) {
  __shared__ float gates[4][16][16];
  const int gate = threadIdx.x >> 5;      // 0..3
  const int j0 = blockIdx.x * 16;         // hidden slice
  const int n0 = gate * kH + j0;          // column offset in [4H] gate space
  f32x8 acc = {};
#pragma unroll 4
  for (int k0 = 0; k0 < kH; k0 += 32) {   // input contribution
    bf16x16 a = load_a(x_t, x_stride, k0);
    bf16x16 b = load_bT(wih, kH, n0, k0);
    acc = wmma_bf16(a, b, acc);
  }
#pragma unroll 4
  for (int k0 = 0; k0 < kH; k0 += 32) {   // recurrent contribution
    bf16x16 a = load_a(h_in, kH, k0);
    bf16x16 b = load_bT(whh, kH, n0, k0);
    acc = wmma_bf16(a, b, acc);
  }
  const int l = lane_id();
  const int jl = l & 15;
  const int mb = (l >> 4) * 8;
  const float bv = bih[n0 + jl] + bhh[n0 + jl];
#pragma unroll
  for (int r = 0; r < 8; ++r) gates[gate][mb + r][jl] = acc[r] + bv;
  __syncthreads();
  for (int e = threadIdx.x; e < 256; e += 128) {
    const int b = e >> 4, j = e & 15;
    const int col = j0 + j;
    const float gi = gates[0][b][j], gf = gates[1][b][j];
    const float gg = gates[2][b][j], go = gates[3][b][j];
    const float si = 1.0f / (1.0f + __expf(-gi));
    const float sf = 1.0f / (1.0f + __expf(-gf));
    const float so = 1.0f / (1.0f + __expf(-go));
    const float cn = sf * c_st[b * kH + col] + si * tanhf(gg);
    const float hn = so * tanhf(cn);
    c_st[b * kH + col] = cn;
    h_out[b * kH + col] = hn;
    y_t[b * y_stride + col] = hn;
  }
}

// ---------------- Kernel 3: FF up-projection + ReLU --------------------------
// out[m,d] = relu(sum_h A[m,h] * W1t[d,h] + b1[d]); W1t pre-transposed bf16 [D,H].
__global__ void ff1_kernel(const float* __restrict__ A, const __bf16* __restrict__ W1t,
                           const float* __restrict__ B1, float* __restrict__ out) {
  const int f = blockIdx.y;
  const int tile = blockIdx.x * 8 + (threadIdx.x >> 5);
  const int NT = kD / 16;                 // 128
  const int mg = tile / NT, nt = tile % NT;   // mg: 64-row group (16 groups)
  const __bf16* Wf = W1t + (size_t)f * kD * kH;
  const float* Am = A + (size_t)mg * 64 * kH;
  f32x8 acc[4] = {};
  for (int k0 = 0; k0 < kH; k0 += 32) {
    bf16x16 b = load_bT_bf16(Wf, kH, nt * 16, k0);
#pragma unroll
    for (int s = 0; s < 4; ++s) {
      bf16x16 a = load_a(Am + (size_t)s * 16 * kH, kH, k0);
      acc[s] = wmma_bf16(a, b, acc[s]);
    }
  }
  float* Of = out + (size_t)f * kM * kD + (size_t)mg * 64 * kD;
#pragma unroll
  for (int s = 0; s < 4; ++s)
    store_d(Of + (size_t)s * 16 * kD, kD, nt * 16, acc[s], B1 + f * kD, true);
}

// ---------------- Kernel 4: FF down-projection -------------------------------
// out[m,h] = sum_d A[m,d] * W2t[h,d] + b2[h]; W2t pre-transposed bf16 [H,D].
__global__ void ff2_kernel(const float* __restrict__ A, const __bf16* __restrict__ W2t,
                           const float* __restrict__ B2, float* __restrict__ out) {
  const int f = blockIdx.y;
  const int tile = blockIdx.x * 8 + (threadIdx.x >> 5);
  const int NT = kH / 16;                 // 32
  const int mg = tile / NT, nt = tile % NT;
  const __bf16* Wf = W2t + (size_t)f * kH * kD;
  const float* Am = A + (size_t)f * kM * kD + (size_t)mg * 64 * kD;
  f32x8 acc[4] = {};
  for (int k0 = 0; k0 < kD; k0 += 32) {
    bf16x16 b = load_bT_bf16(Wf, kD, nt * 16, k0);
#pragma unroll
    for (int s = 0; s < 4; ++s) {
      bf16x16 a = load_a(Am + (size_t)s * 16 * kD, kD, k0);
      acc[s] = wmma_bf16(a, b, acc[s]);
    }
  }
  float* Of = out + (size_t)f * kM * kH + (size_t)mg * 64 * kH;
#pragma unroll
  for (int s = 0; s < 4; ++s)
    store_d(Of + (size_t)s * 16 * kH, kH, nt * 16, acc[s], B2 + f * kH, false);
}

// ---------------- Kernel 5: vocab projection ([M,H] x [V,H]^T per factor) ----
// logits[(m*3 + f)*V + v] = sum_h ffh[f][m][h] * out_w[f][v][h] + out_b[f][v]
// 4 M-tiles per wave; weight fragments fetched once per 64 rows; NT stores.
__global__ void logits_kernel(const float* __restrict__ Aff, const float* __restrict__ Wv,
                              const float* __restrict__ bv, float* __restrict__ out) {
  const int f = blockIdx.y;
  const int tile = blockIdx.x * 8 + (threadIdx.x >> 5);
  const int NT = kV / 16;                 // 2000
  const int mg = tile / NT, nt = tile % NT;   // mg: 0..15 (64-row group)
  const int n0 = nt * 16;
  const float* Af = Aff + (size_t)f * kM * kH + (size_t)mg * 64 * kH;
  const float* Wf = Wv + (size_t)f * kV * kH;
  f32x8 acc[4] = {};
  for (int k0 = 0; k0 < kH; k0 += 32) {
    __builtin_prefetch(Wf + (size_t)(n0 + (threadIdx.x & 15)) * kH + k0 + 64, 0, 1);
    bf16x16 b = load_bT(Wf, kH, n0, k0);
#pragma unroll
    for (int s = 0; s < 4; ++s) {
      bf16x16 a = load_a(Af + (size_t)s * 16 * kH, kH, k0);
      acc[s] = wmma_bf16(a, b, acc[s]);
    }
  }
  const int l = lane_id();
  const int v = n0 + (l & 15);
  const float bb = bv[(size_t)f * kV + v];
#pragma unroll
  for (int s = 0; s < 4; ++s) {
    const int m0 = mg * 64 + s * 16 + ((l >> 4) * 8);
#pragma unroll
    for (int r = 0; r < 8; ++r) {
      // streaming output, never re-read: non-temporal to keep out_w resident in L2
      __builtin_nontemporal_store(acc[s][r] + bb,
                                  &out[((size_t)(m0 + r) * kF + f) * kV + v]);
    }
  }
}

// ---------------- small state init / output copy kernels ---------------------
__global__ void init_state_kernel(const float* __restrict__ h0, const float* __restrict__ c0,
                                  float* __restrict__ hb, float* __restrict__ cb) {
  const int i = blockIdx.x * blockDim.x + threadIdx.x;   // over L*B*H = 16384
  if (i < 2 * kB * kH) {
    const int l = i / (kB * kH);
    const int r = i % (kB * kH);
    hb[(size_t)l * 2 * kB * kH + r] = h0[i];             // ping-pong slot 0
    cb[i] = c0[i];
  }
}

__global__ void copy_hc_kernel(const float* __restrict__ hb, const float* __restrict__ cb,
                               float* __restrict__ out) {
  const int i = blockIdx.x * blockDim.x + threadIdx.x;
  if (i < 2 * kB * kH) {
    const int l = i / (kB * kH);
    const int r = i % (kB * kH);
    out[i] = hb[(size_t)l * 2 * kB * kH + r];            // final h in slot (T%2)==0
    out[2 * kB * kH + i] = cb[i];
  }
}

// ---------------- host launcher ----------------------------------------------
extern "C" void kernel_launch(void* const* d_in, const int* in_sizes, int n_in,
                              void* d_out, int out_size, void* d_ws, size_t ws_size,
                              hipStream_t stream) {
  (void)in_sizes; (void)n_in; (void)out_size; (void)ws_size;
  const float* h0   = (const float*)d_in[1];
  const float* c0   = (const float*)d_in[2];
  const int*   tgt  = (const int*)d_in[3];
  const int*   bos  = (const int*)d_in[4];
  const float* emb  = (const float*)d_in[5];
  const float* epw  = (const float*)d_in[6];
  const float* epb  = (const float*)d_in[7];
  const float* ffw1 = (const float*)d_in[8];
  const float* ffb1 = (const float*)d_in[9];
  const float* ffw2 = (const float*)d_in[10];
  const float* ffb2 = (const float*)d_in[11];
  const float* outw = (const float*)d_in[12];
  const float* outb = (const float*)d_in[13];
  const float* wih[2] = {(const float*)d_in[14], (const float*)d_in[18]};
  const float* whh[2] = {(const float*)d_in[15], (const float*)d_in[19]};
  const float* bih[2] = {(const float*)d_in[16], (const float*)d_in[20]};
  const float* bhh[2] = {(const float*)d_in[17], (const float*)d_in[21]};

  float* ws = (float*)d_ws;
  size_t off = 0;
  float* x    = ws + off; off += (size_t)kM * kH;          // [1024,512]
  float* seq0 = ws + off; off += (size_t)kM * kH;          // layer-0 outputs
  float* seq1 = ws + off; off += (size_t)kM * kH;          // layer-1 outputs ("out")
  float* hb   = ws + off; off += 4 * (size_t)kB * kH;      // [layer][pingpong][B*H]
  float* cb   = ws + off; off += 2 * (size_t)kB * kH;      // [layer][B*H]
  float* ffd  = ws + off; off += (size_t)kF * kM * kD;     // relu'd FF inner
  float* ffh  = ws + off; off += (size_t)kF * kM * kH;     // FF outputs
  __bf16* w1t = (__bf16*)(ws + off); off += (size_t)kF * kD * kH / 2;  // bf16 [F][D][H]
  __bf16* w2t = (__bf16*)(ws + off); off += (size_t)kF * kH * kD / 2;  // bf16 [F][H][D]

  // one-time weight prep: K-major fp32 weights -> N-major bf16 (contiguous B frags)
  transpose_to_bf16_kernel<<<dim3(4096, kF), 256, 0, stream>>>(ffw1, w1t, kH, kD);
  transpose_to_bf16_kernel<<<dim3(4096, kF), 256, 0, stream>>>(ffw2, w2t, kD, kH);

  init_state_kernel<<<64, 256, 0, stream>>>(h0, c0, hb, cb);
  embed_proj_kernel<<<256, 256, 0, stream>>>(tgt, bos, emb, epw, epb, x);

  for (int t = 0; t < kT; ++t) {
    for (int lyr = 0; lyr < 2; ++lyr) {
      float* hin  = hb + (size_t)lyr * 2 * kB * kH + (size_t)(t & 1) * kB * kH;
      float* hout = hb + (size_t)lyr * 2 * kB * kH + (size_t)((t + 1) & 1) * kB * kH;
      float* cst  = cb + (size_t)lyr * kB * kH;
      const float* xin = (lyr == 0) ? (x + (size_t)t * kH) : (seq0 + (size_t)t * kH);
      float* y         = (lyr == 0) ? (seq0 + (size_t)t * kH) : (seq1 + (size_t)t * kH);
      lstm_step_kernel<<<kH / 16, 128, 0, stream>>>(xin, kT * kH, wih[lyr], whh[lyr],
                                                    bih[lyr], bhh[lyr], hin, hout, cst,
                                                    y, kT * kH);
    }
  }

  // tiles = (M/64 groups) * N-tiles; 8 waves per block
  ff1_kernel<<<dim3((kM / 64) * (kD / 16) / 8, kF), 256, 0, stream>>>(seq1, w1t, ffb1, ffd);
  ff2_kernel<<<dim3((kM / 64) * (kH / 16) / 8, kF), 256, 0, stream>>>(ffd, w2t, ffb2, ffh);
  logits_kernel<<<dim3((kM / 64) * (kV / 16) / 8, kF), 256, 0, stream>>>(ffh, outw, outb,
                                                                         (float*)d_out);
  copy_hc_kernel<<<64, 256, 0, stream>>>(hb, cb, (float*)d_out + (size_t)kM * kF * kV);
}
